// CDSPMoELayer_87101936763275
// MI455X (gfx1250) — matrix operand order repaired and checked by reading, not compiled, not run on hardware
//
#include <hip/hip_runtime.h>
#include <hip/hip_bf16.h>

// ---------------------------------------------------------------------------
// CDSP MoE layer for MI455X (gfx1250), wave32 + v_wmma_f32_16x16x32_bf16.
// Shapes (fixed by setup_inputs): B=2 S=4096 D=1024 E=16 DT=32 DB=256 R=64 K=2
// ---------------------------------------------------------------------------

typedef __bf16 v16bf __attribute__((ext_vector_type(16)));
typedef __bf16 v8bf  __attribute__((ext_vector_type(8)));
typedef float  v8f   __attribute__((ext_vector_type(8)));

#define D_MODEL 1024
#define S_SEQ   4096
#define NB      2
#define N_TOK   (NB * S_SEQ)      // 8192
#define NE      16
#define DTASK   32
#define DBOT    256
#define RQ      64
#define EPSLN   1e-5f
#define LOG2E   1.4426950408889634f

// ---- workspace byte offsets (all 256B+ aligned) ----
#define OFF_PART   0ull            // 1024 floats: psum[2][256], psq[2][256]
#define OFF_MU     4096ull         // 4 floats: mu0,rstd0,mu1,rstd1
#define OFF_CB     5120ull         // 32 floats: per (batch,expert) router constant
#define OFF_GATE   6144ull         // 16 floats
#define OFF_TLIDX  8192ull         // 1024 ints: tl_idx[E][R]
#define OFF_SEL    16384ull        // 16384 ints: expert per slot (slot = 2n+k)
#define OFF_COMBW  81920ull        // 16384 floats: w*gate per slot
#define OFF_LIST   147456ull       // 16384 ints: slots grouped by expert
#define OFF_CNT    212992ull       // 16 ints
#define OFF_EOFF   214016ull       // 16 ints
#define OFF_WDG    (1ull << 20)    // 2 MB bf16: Wd gathered [E][R][D]
#define OFF_WUT    (3ull << 20)    // 2 MB bf16: Wu gathered+transposed [E][D][R]
#define OFF_YP     (5ull << 20)    // 64 MB + 4 KB: partial planes [2][N][D] + dump row

#define WMMA_BF16(a, b, c) \
    __builtin_amdgcn_wmma_f32_16x16x32_bf16(false, (a), false, (b), (short)0, (c), false, false)

// ---------------------------------------------------------------------------
// 1) LayerNorm stats, stage 1: fixed-order partial sums (deterministic)
// ---------------------------------------------------------------------------
__global__ __launch_bounds__(256) void k_stats_partial(const float* __restrict__ x,
                                                       float* __restrict__ part)
{
    int b = blockIdx.y, blk = blockIdx.x, t = threadIdx.x;
    const float* xb = x + (size_t)b * S_SEQ * D_MODEL;
    size_t base = (size_t)blk * 16384;
    float s = 0.f, sq = 0.f;
    for (int j = 0; j < 64; ++j) {
        float v = xb[base + t + j * 256];
        s += v; sq += v * v;
    }
    __shared__ float r0[256], r1[256];
    r0[t] = s; r1[t] = sq; __syncthreads();
    for (int st = 128; st > 0; st >>= 1) {
        if (t < st) { r0[t] += r0[t + st]; r1[t] += r1[t + st]; }
        __syncthreads();
    }
    if (t == 0) { part[b * 256 + blk] = r0[0]; part[512 + b * 256 + blk] = r1[0]; }
}

// ---------------------------------------------------------------------------
// 2) Finalize mu/rstd + fold bias, task term, and -rstd*mu*colsum(Wr) into cb
// ---------------------------------------------------------------------------
__global__ __launch_bounds__(256) void k_finalize(const float* __restrict__ part,
                                                  const int*   __restrict__ task_id,
                                                  const float* __restrict__ task_emb,
                                                  const float* __restrict__ Wr,
                                                  const float* __restrict__ br,
                                                  float* __restrict__ muRstd,
                                                  float* __restrict__ cb)
{
    int t = threadIdx.x;
    __shared__ float red[256];
    __shared__ float mu_sh[NB], rs_sh[NB];
    for (int b = 0; b < NB; ++b) {
        red[t] = part[b * 256 + t]; __syncthreads();
        for (int st = 128; st > 0; st >>= 1) { if (t < st) red[t] += red[t + st]; __syncthreads(); }
        float sum = red[0]; __syncthreads();
        red[t] = part[512 + b * 256 + t]; __syncthreads();
        for (int st = 128; st > 0; st >>= 1) { if (t < st) red[t] += red[t + st]; __syncthreads(); }
        float sq = red[0]; __syncthreads();
        if (t == 0) {
            const float M = (float)((size_t)S_SEQ * D_MODEL);
            float mu  = sum / M;
            float var = sq / M - mu * mu;
            float rs  = rsqrtf(var + EPSLN);
            mu_sh[b] = mu; rs_sh[b] = rs;
            muRstd[b * 2] = mu; muRstd[b * 2 + 1] = rs;
        }
        __syncthreads();
    }
    if (t < NB * NE) {
        int b = t >> 4, e = t & 15;
        float cs = 0.f;
        for (int d = 0; d < D_MODEL; ++d) cs += Wr[d * NE + e];
        int tid = task_id[b];
        float tk = 0.f;
        for (int j = 0; j < DTASK; ++j)
            tk += task_emb[tid * DTASK + j] * Wr[(D_MODEL + j) * NE + e];
        cb[b * NE + e] = br[e] + tk - rs_sh[b] * mu_sh[b] * cs;
    }
}

// ---------------------------------------------------------------------------
// 3) Topology: top-64 per expert by rank-counting (stable ties like top_k),
//    gate[e] = mean(sigmoid(top vals)); deterministic tree reduction.
// ---------------------------------------------------------------------------
__global__ __launch_bounds__(256) void k_topo(const float* __restrict__ topo,
                                              float* __restrict__ gate,
                                              int*   __restrict__ tlidx)
{
    int t = threadIdx.x;
    __shared__ float row[256];
    __shared__ float sr[256];
    for (int e = 0; e < NE; ++e) {
        row[t] = topo[e * DBOT + t];
        __syncthreads();
        float v = row[t];
        int rank = 0;
        for (int j = 0; j < DBOT; ++j) {
            float vj = row[j];
            rank += (vj > v) || (vj == v && j < t);
        }
        float contrib = 0.f;
        if (rank < RQ) {
            tlidx[e * RQ + rank] = t;
            contrib = __builtin_amdgcn_rcpf(1.f + __builtin_amdgcn_exp2f(-LOG2E * v));
        }
        sr[t] = contrib; __syncthreads();
        for (int st = 128; st > 0; st >>= 1) { if (t < st) sr[t] += sr[t + st]; __syncthreads(); }
        if (t == 0) gate[e] = sr[0] / (float)RQ;
        __syncthreads();
    }
}

// ---------------------------------------------------------------------------
// 4) Gather expert bases: Wdg[e][r][d] = W_down[d][idx] (bf16, K-contiguous),
//    Wut[e][d][r] = W_up[idx][d]      (bf16, transposed so GEMM2 B is contiguous)
// ---------------------------------------------------------------------------
__global__ __launch_bounds__(256) void k_gather(const float* __restrict__ Wdown,
                                                const float* __restrict__ Wup,
                                                const int*   __restrict__ tlidx,
                                                __bf16* __restrict__ Wdg,
                                                __bf16* __restrict__ Wut)
{
    int er = blockIdx.x;            // e*64 + r
    int e = er >> 6, r = er & 63;
    int idx = tlidx[er];
    for (int d = threadIdx.x; d < D_MODEL; d += 256) {
        Wdg[(size_t)er * D_MODEL + d]           = (__bf16)Wdown[(size_t)d * DBOT + idx];
        Wut[((size_t)e * D_MODEL + d) * RQ + r] = (__bf16)Wup[(size_t)idx * D_MODEL + d];
    }
}

// ---------------------------------------------------------------------------
// 5) Router: logits = rstd*dot(x,Wr_col) + cb; per-token top-2 + softmax*gate
// ---------------------------------------------------------------------------
__global__ __launch_bounds__(256) void k_router(const float* __restrict__ x,
                                                const float* __restrict__ Wr,
                                                const float* __restrict__ muRstd,
                                                const float* __restrict__ cb,
                                                const float* __restrict__ gate,
                                                int*   __restrict__ sel,
                                                float* __restrict__ combw)
{
    __shared__ float lg[16][16];
    int t = threadIdx.x;
    int tok0 = blockIdx.x * 16;
    int tk = t >> 4, e = t & 15;
    int n = tok0 + tk;
    const float* xr = x + (size_t)n * D_MODEL;
    float acc = 0.f;
    for (int d = 0; d < D_MODEL; ++d) acc += xr[d] * Wr[d * NE + e];
    int b = n >> 12;                      // n / 4096
    lg[tk][e] = muRstd[b * 2 + 1] * acc + cb[b * NE + e];
    __syncthreads();
    if (t < 16) {
        int nn = tok0 + t;
        float b1 = -1e30f; int i1 = 0;
        for (int j = 0; j < NE; ++j) { float v = lg[t][j]; if (v > b1) { b1 = v; i1 = j; } }
        float b2 = -1e30f; int i2 = 0;
        for (int j = 0; j < NE; ++j) { if (j == i1) continue; float v = lg[t][j]; if (v > b2) { b2 = v; i2 = j; } }
        float e2 = __builtin_amdgcn_exp2f(LOG2E * (b2 - b1));   // arg <= 0, branch-free
        float inv = __builtin_amdgcn_rcpf(1.f + e2);
        float w1 = inv;
        float w2 = e2 * inv;
        sel[nn * 2]     = i1; combw[nn * 2]     = w1 * gate[i1];
        sel[nn * 2 + 1] = i2; combw[nn * 2 + 1] = w2 * gate[i2];
    }
}

// ---------------------------------------------------------------------------
// 6) Deterministic expert lists: block e scans all 2N slots in slot order,
//    offset = #slots with sel<e; stable append via ballot_w32 prefix.
// ---------------------------------------------------------------------------
__global__ __launch_bounds__(256) void k_list(const int* __restrict__ sel,
                                              int* __restrict__ list,
                                              int* __restrict__ counts,
                                              int* __restrict__ eoff)
{
    int e = blockIdx.x, t = threadIdx.x;
    int lane = t & 31, w = t >> 5;
    __shared__ int red[256];
    __shared__ int wsum[8];
    int less = 0;
    for (int i = t; i < 2 * N_TOK; i += 256) less += (sel[i] < e);
    red[t] = less; __syncthreads();
    for (int st = 128; st > 0; st >>= 1) { if (t < st) red[t] += red[t + st]; __syncthreads(); }
    int off = red[0];
    __syncthreads();

    unsigned lmask = (1u << lane) - 1u;
    int cursor = 0;
    for (int base = 0; base < 2 * N_TOK; base += 256) {
        int s = base + t;
        int flag = (sel[s] == e) ? 1 : 0;
        unsigned msk = __builtin_amdgcn_ballot_w32(flag != 0);
        if (lane == 0) wsum[w] = __builtin_popcount(msk);
        __syncthreads();
        int pre = 0, total = 0;
        #pragma unroll
        for (int j = 0; j < 8; ++j) { int v = wsum[j]; total += v; if (j < w) pre += v; }
        if (flag) list[off + cursor + pre + __builtin_popcount(msk & lmask)] = s;
        cursor += total;
        __syncthreads();
    }
    if (t == 0) { counts[e] = cursor; eoff[e] = off; }
}

// ---------------------------------------------------------------------------
// 7) Fused expert kernel: one wave handles a 32-token tile of one expert.
//    GEMM1: H[32 x 64]  = X[32 x 1024] @ Wd[e]^T   (ping-pong pipeline, 8 WMMA/k)
//    abar  = gelu_tanh(H) * combweight -> LDS (C-frag -> A-frag relayout)
//    GEMM2: Y[32 x 1024] = abar @ Wu[e]            (ping-pong pipeline, 4 WMMA/dt)
//    Pad rows stream into a dump row -> branch-free stores, deterministic.
// ---------------------------------------------------------------------------
__device__ __forceinline__ float gelu_fast(float u)
{
    // gelu_tanh(u) = u * sigmoid(2*z), z = sqrt(2/pi)*(u + 0.044715 u^3)
    float z = 0.7978845608028654f * (u + 0.044715f * u * u * u);
    return u * __builtin_amdgcn_rcpf(1.f + __builtin_amdgcn_exp2f(-2.f * LOG2E * z));
}

__device__ __forceinline__ v16bf load_a_frag(const float* __restrict__ xr, int d0)
{
    const float4 f0 = *(const float4*)(xr + d0);
    const float4 f1 = *(const float4*)(xr + d0 + 4);
    const float4 f2 = *(const float4*)(xr + d0 + 8);
    const float4 f3 = *(const float4*)(xr + d0 + 12);
    v16bf a;
    a[0]  = (__bf16)f0.x; a[1]  = (__bf16)f0.y; a[2]  = (__bf16)f0.z; a[3]  = (__bf16)f0.w;
    a[4]  = (__bf16)f1.x; a[5]  = (__bf16)f1.y; a[6]  = (__bf16)f1.z; a[7]  = (__bf16)f1.w;
    a[8]  = (__bf16)f2.x; a[9]  = (__bf16)f2.y; a[10] = (__bf16)f2.z; a[11] = (__bf16)f2.w;
    a[12] = (__bf16)f3.x; a[13] = (__bf16)f3.y; a[14] = (__bf16)f3.z; a[15] = (__bf16)f3.w;
    return a;
}

__device__ __forceinline__ v16bf load_a_lds(const __bf16* rowp, int kb, int h)
{
    const v8bf lo = *(const v8bf*)(rowp + kb * 32 + 16 * h);
    const v8bf hi = *(const v8bf*)(rowp + kb * 32 + 16 * h + 8);
    return __builtin_shufflevector(lo, hi, 0, 1, 2, 3, 4, 5, 6, 7,
                                           8, 9, 10, 11, 12, 13, 14, 15);
}

__global__ __launch_bounds__(32) void k_expert(const float* __restrict__ x,
                                               const int*   __restrict__ list,
                                               const int*   __restrict__ counts,
                                               const int*   __restrict__ eoff,
                                               const float* __restrict__ combw,
                                               const __bf16* __restrict__ Wdg,
                                               const __bf16* __restrict__ Wut,
                                               float* __restrict__ yp)
{
    int e = blockIdx.y;
    int tile = blockIdx.x;
    int cnt = counts[e];
    if (tile * 32 >= cnt) return;
    int base = eoff[e] + tile * 32;

    __shared__ float    c_sh[32];
    __shared__ int      n_sh[32];
    __shared__ unsigned o_sh[32];
    __shared__ __attribute__((aligned(32))) __bf16 ab[32][64];

    int lane = threadIdx.x;
    {
        int valid = (tile * 32 + lane) < cnt;
        int s = valid ? list[base + lane] : 0;
        n_sh[lane] = valid ? (s >> 1) : 0;
        c_sh[lane] = valid ? combw[s] : 0.f;
        o_sh[lane] = valid ? (unsigned)(((s & 1) * N_TOK + (s >> 1)) * D_MODEL)
                           : (unsigned)(2u * N_TOK * D_MODEL);   // dump row
    }
    __syncthreads();

    int m = lane & 15;          // A row-in-tile / B,C column per ISA layout
    int h = lane >> 4;          // lane-half selector
    const float*  xr0 = x + (size_t)n_sh[m]      * D_MODEL;
    const float*  xr1 = x + (size_t)n_sh[16 + m] * D_MODEL;
    const __bf16* wd  = Wdg + (size_t)e * RQ * D_MODEL;
    const __bf16* wu  = Wut + (size_t)e * D_MODEL * RQ;

    // warm L2 with this expert's GEMM2 weights while GEMM1 runs
    __builtin_prefetch(wu + (size_t)lane * 2048, 0, 1);

    const v8f vz = {0.f, 0.f, 0.f, 0.f, 0.f, 0.f, 0.f, 0.f};
    v8f acc[8];
    #pragma unroll
    for (int i = 0; i < 8; ++i) acc[i] = vz;

    // B frag (nt, kb): lane holds col r = nt*16+m, K = kb*32 + 16h + j
    const __bf16* wdm = wd + (size_t)m * D_MODEL + 16 * h;
#define LDB1(nt, kb) (*(const v16bf*)(wdm + (size_t)(nt) * 16 * D_MODEL + (size_t)(kb) * 32))

    // ---- GEMM1: ping-pong pipeline (no register rotation copies) ----
    v16bf a0A, a1A, b0A, b1A, b2A, b3A;
    v16bf a0B, a1B, b0B, b1B, b2B, b3B;

#define G1_LOAD(sfx, kb) do {                                   \
        a0##sfx = load_a_frag(xr0, (kb) * 32 + 16 * h);         \
        a1##sfx = load_a_frag(xr1, (kb) * 32 + 16 * h);         \
        b0##sfx = LDB1(0, (kb)); b1##sfx = LDB1(1, (kb));       \
        b2##sfx = LDB1(2, (kb)); b3##sfx = LDB1(3, (kb));       \
    } while (0)

#define G1_COMPUTE(sfx) do {                                    \
        acc[0] = WMMA_BF16(a0##sfx, b0##sfx, acc[0]);           \
        acc[1] = WMMA_BF16(a0##sfx, b1##sfx, acc[1]);           \
        acc[2] = WMMA_BF16(a0##sfx, b2##sfx, acc[2]);           \
        acc[3] = WMMA_BF16(a0##sfx, b3##sfx, acc[3]);           \
        acc[4] = WMMA_BF16(a1##sfx, b0##sfx, acc[4]);           \
        acc[5] = WMMA_BF16(a1##sfx, b1##sfx, acc[5]);           \
        acc[6] = WMMA_BF16(a1##sfx, b2##sfx, acc[6]);           \
        acc[7] = WMMA_BF16(a1##sfx, b3##sfx, acc[7]);           \
    } while (0)

    G1_LOAD(A, 0);
    for (int kbp = 0; kbp < 16; ++kbp) {
        int k1 = 2 * kbp + 1;
        int k2 = (2 * kbp + 2 < 32) ? 2 * kbp + 2 : 31;   // clamped tail reload
        G1_LOAD(B, k1);
        G1_COMPUTE(A);      // kb = 2*kbp, covered by stage-B loads above
        G1_LOAD(A, k2);
        G1_COMPUTE(B);      // kb = 2*kbp+1, covered by stage-A loads above
    }

    // ---- gelu * comb, relayout C-frag -> row-major LDS (bf16) ----
    #pragma unroll
    for (int mt = 0; mt < 2; ++mt) {
        #pragma unroll
        for (int nt = 0; nt < 4; ++nt) {
            #pragma unroll
            for (int g = 0; g < 8; ++g) {
                int mrow = mt * 16 + g + 8 * h;        // C layout: M = g + 8h, N = m
                float av = gelu_fast(acc[mt * 4 + nt][g]) * c_sh[mrow];
                ab[mrow][nt * 16 + m] = (__bf16)av;
            }
        }
    }
    __syncthreads();

    // ---- GEMM2: A fragments from LDS, ping-pong B over 64 d-tiles ----
    v16bf a20 = load_a_lds(&ab[m][0],      0, h);
    v16bf a21 = load_a_lds(&ab[m][0],      1, h);
    v16bf a30 = load_a_lds(&ab[16 + m][0], 0, h);
    v16bf a31 = load_a_lds(&ab[16 + m][0], 1, h);

    unsigned ob0[8], ob1[8];
    #pragma unroll
    for (int g = 0; g < 8; ++g) { ob0[g] = o_sh[g + 8 * h]; ob1[g] = o_sh[16 + g + 8 * h]; }

    // B frag (dt, kb): lane holds col d = dt*16+m, r = kb*32 + 16h + j
    const __bf16* wum = wu + (size_t)m * RQ + 16 * h;
#define LDB2(dt, kb) (*(const v16bf*)(wum + (size_t)(dt) * 16 * RQ + (size_t)(kb) * 32))

    v16bf bu0A, bu1A, bu0B, bu1B;

#define G2_LOAD(sfx, dt) do {                                   \
        bu0##sfx = LDB2((dt), 0); bu1##sfx = LDB2((dt), 1);     \
    } while (0)

#define G2_COMPUTE_STORE(sfx, dt) do {                          \
        v8f y0 = vz, y1 = vz;                                   \
        y0 = WMMA_BF16(a20, bu0##sfx, y0);                      \
        y0 = WMMA_BF16(a21, bu1##sfx, y0);                      \
        y1 = WMMA_BF16(a30, bu0##sfx, y1);                      \
        y1 = WMMA_BF16(a31, bu1##sfx, y1);                      \
        int dcol = (dt) * 16 + m;                               \
        _Pragma("unroll")                                       \
        for (int g = 0; g < 8; ++g) {                           \
            yp[(size_t)ob0[g] + dcol] = y0[g];                  \
            yp[(size_t)ob1[g] + dcol] = y1[g];                  \
        }                                                       \
    } while (0)

    G2_LOAD(A, 0);
    for (int dtp = 0; dtp < 32; ++dtp) {
        int d1 = 2 * dtp + 1;
        int d2 = (2 * dtp + 2 < 64) ? 2 * dtp + 2 : 63;   // clamped tail reload
        G2_LOAD(B, d1);
        G2_COMPUTE_STORE(A, 2 * dtp);
        G2_LOAD(A, d2);
        G2_COMPUTE_STORE(B, d1);
    }
#undef LDB1
#undef LDB2
#undef G1_LOAD
#undef G1_COMPUTE
#undef G2_LOAD
#undef G2_COMPUTE_STORE
}

// ---------------------------------------------------------------------------
// 8) Combine the two per-slot planes -> y
// ---------------------------------------------------------------------------
__global__ __launch_bounds__(256) void k_combine(const float* __restrict__ yp,
                                                 float* __restrict__ out)
{
    size_t i = (size_t)blockIdx.x * 256 + threadIdx.x;   // over float4s
    const float4* a = (const float4*)yp;
    const float4* b = (const float4*)(yp + (size_t)N_TOK * D_MODEL);
    float4 va = a[i], vb = b[i];
    float4 r; r.x = va.x + vb.x; r.y = va.y + vb.y; r.z = va.z + vb.z; r.w = va.w + vb.w;
    ((float4*)out)[i] = r;
}

// ---------------------------------------------------------------------------
extern "C" void kernel_launch(void* const* d_in, const int* in_sizes, int n_in,
                              void* d_out, int out_size, void* d_ws, size_t ws_size,
                              hipStream_t stream)
{
    (void)in_sizes; (void)n_in; (void)out_size; (void)ws_size;

    const float* x        = (const float*)d_in[0];
    const int*   task_id  = (const int*)  d_in[1];
    const float* task_emb = (const float*)d_in[2];
    const float* Wr       = (const float*)d_in[3];
    const float* br       = (const float*)d_in[4];
    const float* Wdown    = (const float*)d_in[5];
    const float* Wup      = (const float*)d_in[6];
    const float* topo     = (const float*)d_in[7];
    float* out = (float*)d_out;

    char* ws = (char*)d_ws;
    float*  part   = (float*)(ws + OFF_PART);
    float*  muRstd = (float*)(ws + OFF_MU);
    float*  cbp    = (float*)(ws + OFF_CB);
    float*  gate   = (float*)(ws + OFF_GATE);
    int*    tlidx  = (int*)  (ws + OFF_TLIDX);
    int*    sel    = (int*)  (ws + OFF_SEL);
    float*  combw  = (float*)(ws + OFF_COMBW);
    int*    list   = (int*)  (ws + OFF_LIST);
    int*    counts = (int*)  (ws + OFF_CNT);
    int*    eoff   = (int*)  (ws + OFF_EOFF);
    __bf16* Wdg    = (__bf16*)(ws + OFF_WDG);
    __bf16* Wut    = (__bf16*)(ws + OFF_WUT);
    float*  yp     = (float*)(ws + OFF_YP);

    k_stats_partial<<<dim3(256, NB), 256, 0, stream>>>(x, part);
    k_finalize<<<1, 256, 0, stream>>>(part, task_id, task_emb, Wr, br, muRstd, cbp);
    k_topo<<<1, 256, 0, stream>>>(topo, gate, tlidx);
    k_gather<<<NE * RQ, 256, 0, stream>>>(Wdown, Wup, tlidx, Wdg, Wut);
    k_router<<<N_TOK / 16, 256, 0, stream>>>(x, Wr, muRstd, cbp, gate, sel, combw);
    k_list<<<NE, 256, 0, stream>>>(sel, list, counts, eoff);
    k_expert<<<dim3(256, NE), 32, 0, stream>>>(x, list, counts, eoff, combw, Wdg, Wut, yp);
    k_combine<<<(N_TOK * D_MODEL) / (256 * 4), 256, 0, stream>>>(yp, out);
}
